// DocREModel_KD_20770461843714
// MI455X (gfx1250) — matrix operand order repaired and checked by reading, not compiled, not run on hardware
//
#include <hip/hip_runtime.h>
#include <hip/hip_bf16.h>

typedef __attribute__((ext_vector_type(16))) _Float16 v16h;
typedef __attribute__((ext_vector_type(8)))  _Float16 v8h;
typedef __attribute__((ext_vector_type(8)))  float    v8f;

// Problem constants (from reference)
#define NB   4
#define NH   12
#define NC   1024
#define NHS  768
#define NE   42
#define NM   4
#define NR   (NE * NE)       // 1764
#define MT   111             // ceil(1764/16)
#define NT64 12              // 768/64 : one wave covers 64 output columns

// ---------------------------------------------------------------------------
// K1: e_emb[b,e,d] = logsumexp_m seq[b, starts[b,e,m]+1, d]
// ---------------------------------------------------------------------------
__global__ void k_e_emb(const float* __restrict__ seq, const int* __restrict__ starts,
                        float* __restrict__ eemb) {
  const int b = blockIdx.x / NE;
  const int e = blockIdx.x % NE;
  const int* st = starts + ((size_t)b * NE + e) * NM;
  const int i0 = st[0] + 1, i1 = st[1] + 1, i2 = st[2] + 1, i3 = st[3] + 1;
  const float* sb = seq + (size_t)b * NC * NHS;
  float* o = eemb + ((size_t)b * NE + e) * NHS;
#pragma unroll
  for (int q = 0; q < 3; ++q) {
    const int d = threadIdx.x + q * 256;
    const float x0 = sb[(size_t)i0 * NHS + d];
    const float x1 = sb[(size_t)i1 * NHS + d];
    const float x2 = sb[(size_t)i2 * NHS + d];
    const float x3 = sb[(size_t)i3 * NHS + d];
    const float m = fmaxf(fmaxf(x0, x1), fmaxf(x2, x3));
    const float s = __expf(x0 - m) + __expf(x1 - m) + __expf(x2 - m) + __expf(x3 - m);
    o[d] = m + __logf(s);
  }
}

// ---------------------------------------------------------------------------
// K1b: seqT[b,d,l] = (f16) seq[b,l,d]  — contiguous-K B-matrix tiles for WMMA
// ---------------------------------------------------------------------------
__global__ void k_seqT(const float* __restrict__ seq, _Float16* __restrict__ seqT) {
  const size_t i = (size_t)blockIdx.x * blockDim.x + threadIdx.x; // B*768*1024
  const int l = (int)(i % NC);
  const size_t t = i / NC;
  const int d = (int)(t % NHS);
  const int b = (int)(t / NHS);
  seqT[i] = (_Float16)seq[((size_t)b * NC + l) * NHS + d];
}

// ---------------------------------------------------------------------------
// K2: e_att[b,e,h,c] = mean_m att[b,h,starts[b,e,m]+1,c]
// ---------------------------------------------------------------------------
__global__ void k_e_att(const float* __restrict__ att, const int* __restrict__ starts,
                        float* __restrict__ eatt) {
  const int b = blockIdx.x / (NE * NH);
  const int rem = blockIdx.x % (NE * NH);
  const int e = rem / NH, h = rem % NH;
  const int* st = starts + ((size_t)b * NE + e) * NM;
  const int i0 = st[0] + 1, i1 = st[1] + 1, i2 = st[2] + 1, i3 = st[3] + 1;
  const float* base = att + ((size_t)b * NH + h) * NC * NC;
  float* o = eatt + (((size_t)b * NE + e) * NH + h) * NC;
#pragma unroll
  for (int q = 0; q < 4; ++q) {
    const int c = threadIdx.x + q * 256;
    o[c] = 0.25f * (base[(size_t)i0 * NC + c] + base[(size_t)i1 * NC + c] +
                    base[(size_t)i2 * NC + c] + base[(size_t)i3 * NC + c]);
  }
}

// ---------------------------------------------------------------------------
// K3: ht_att row r: relu(sum_h e_att[hi,h,:]*e_att[ti,h,:]), normalized, -> f16
// ---------------------------------------------------------------------------
__global__ void k_ht_att(const float* __restrict__ eatt, const int* __restrict__ hts,
                         _Float16* __restrict__ A) {
  const int b = blockIdx.x / NR;
  const int r = blockIdx.x % NR;
  const int hi = hts[((size_t)b * NR + r) * 2 + 0];
  const int ti = hts[((size_t)b * NR + r) * 2 + 1];
  const float* H = eatt + ((size_t)b * NE + hi) * NH * NC;
  const float* T = eatt + ((size_t)b * NE + ti) * NH * NC;
  __shared__ float red[256];
  float v[4];
  float local = 0.f;
#pragma unroll
  for (int q = 0; q < 4; ++q) {
    const int c = threadIdx.x + q * 256;
    float s = 0.f;
#pragma unroll
    for (int h = 0; h < NH; ++h) s += H[h * NC + c] * T[h * NC + c];
    s = fmaxf(s, 0.f);
    v[q] = s;
    local += s;
  }
  red[threadIdx.x] = local;
  __syncthreads();
  for (int off = 128; off > 0; off >>= 1) {
    if (threadIdx.x < off) red[threadIdx.x] += red[threadIdx.x + off];
    __syncthreads();
  }
  const float inv = 1.0f / (red[0] + 1e-10f);
  _Float16* Arow = A + ((size_t)b * NR + r) * NC;
#pragma unroll
  for (int q = 0; q < 4; ++q) {
    const int c = threadIdx.x + q * 256;
    Arow[c] = (_Float16)(v[q] * inv);
  }
}

// ---------------------------------------------------------------------------
// K4: rs[b] = ht_att[b] (1764x1024 f16) @ seq[b] (1024x768 f16), f32 accum.
// Register-blocked: one wave computes a 16x64 output tile = 4 WMMA
// accumulators sharing a single A fragment per k-step. K-loop fully unrolled
// so the scheduler software-pipelines loads ahead of WMMAs (partial
// s_wait_loadcnt) instead of serializing with loadcnt==0 per WMMA.
// A 16x32 f16 layout: lane half 0 -> K {k0..k0+15},
// lane half 1 -> K {k0+16..k0+31} (same contiguous order as B); row = lane%16.
// B 32x16 f16: lane holds one column, K = k0+16*half .. +15 (contiguous via
// the seqT pre-transpose; adjacent N-tiles are 16*1024 halves apart).
// C/D: VGPR j -> M = j + 8*(lane/16), N = lane%16.
// ---------------------------------------------------------------------------
__global__ void k_gemm_rs(const _Float16* __restrict__ A,    // (B,1764,1024)
                          const _Float16* __restrict__ Bt,   // (B,768,1024)
                          float* __restrict__ out) {         // rss (B,1764,768)
  const int wave = blockIdx.x * (blockDim.x >> 5) + (threadIdx.x >> 5);
  const int lane = threadIdx.x & 31;
  const int tilesPerDoc = MT * NT64;
  const int b = wave / tilesPerDoc;
  const int t = wave % tilesPerDoc;
  const int m0 = (t / NT64) * 16;
  const int n0 = (t % NT64) * 64;
  const int half = lane >> 4;
  const int l16 = lane & 15;

  int arow = m0 + l16;
  if (arow > NR - 1) arow = NR - 1;                 // clamp; stores are guarded
  const _Float16* Ab = A  + ((size_t)b * NR  + arow)       * NC;
  const _Float16* Bb = Bt + ((size_t)b * NHS + (n0 + l16)) * NC;  // column n0+l16

  v8f c0 = {}, c1 = {}, c2 = {}, c3 = {};
#pragma unroll
  for (int k0 = 0; k0 < NC; k0 += 32) {
    if (k0 + 64 <= NC) {                            // speculative L2 prefetch
      __builtin_prefetch(Ab + k0 + 32, 0, 1);
      __builtin_prefetch(Bb + k0 + 32, 0, 1);
    }
    // A fragment (shared by all 4 n-tiles)
    const v8h alo = *(const v8h*)(Ab + k0 + 16 * half);
    const v8h ahi = *(const v8h*)(Ab + k0 + 16 * half + 8);
    v16h a;
#pragma unroll
    for (int i = 0; i < 8; ++i) { a[i] = alo[i]; a[i + 8] = ahi[i]; }

    // 4 B fragments: n-tiles at column strides of 16 rows of seqT (16*NC halves)
    v16h bm[4];
#pragma unroll
    for (int j = 0; j < 4; ++j) {
      const _Float16* Bj = Bb + (size_t)j * 16 * NC + k0 + 16 * half;
      const v8h blo = *(const v8h*)(Bj);
      const v8h bhi = *(const v8h*)(Bj + 8);
#pragma unroll
      for (int i = 0; i < 8; ++i) { bm[j][i] = blo[i]; bm[j][i + 8] = bhi[i]; }
    }

    c0 = __builtin_amdgcn_wmma_f32_16x16x32_f16(false, a, false, bm[0], (short)0, c0, false, false);
    c1 = __builtin_amdgcn_wmma_f32_16x16x32_f16(false, a, false, bm[1], (short)0, c1, false, false);
    c2 = __builtin_amdgcn_wmma_f32_16x16x32_f16(false, a, false, bm[2], (short)0, c2, false, false);
    c3 = __builtin_amdgcn_wmma_f32_16x16x32_f16(false, a, false, bm[3], (short)0, c3, false, false);
  }

  float* ob = out + (size_t)b * NR * NHS + n0 + l16;
  v8f acc[4] = {c0, c1, c2, c3};
  if (m0 + 16 <= NR) {
    // interior tile (110 of 111 M-tiles): unguarded stores, no exec toggling
#pragma unroll
    for (int j = 0; j < 4; ++j) {
#pragma unroll
      for (int q = 0; q < 8; ++q) {
        const int m = m0 + q + 8 * half;
        ob[(size_t)m * NHS + 16 * j] = acc[j][q];
      }
    }
  } else {
#pragma unroll
    for (int j = 0; j < 4; ++j) {
#pragma unroll
      for (int q = 0; q < 8; ++q) {
        const int m = m0 + q + 8 * half;
        if (m < NR) ob[(size_t)m * NHS + 16 * j] = acc[j][q];
      }
    }
  }
}

// ---------------------------------------------------------------------------
// K5: hss/tss gathers of e_emb by pair indices
// ---------------------------------------------------------------------------
__global__ void k_gather_ht(const float* __restrict__ eemb, const int* __restrict__ hts,
                            float* __restrict__ hss, float* __restrict__ tss) {
  const size_t i = (size_t)blockIdx.x * blockDim.x + threadIdx.x; // B*1764*768
  const int d = (int)(i % NHS);
  const size_t t = i / NHS;
  const int r = (int)(t % NR);
  const int b = (int)(t / NR);
  const int hi = hts[((size_t)b * NR + r) * 2 + 0];
  const int ti = hts[((size_t)b * NR + r) * 2 + 1];
  hss[i] = eemb[((size_t)b * NE + hi) * NHS + d];
  tss[i] = eemb[((size_t)b * NE + ti) * NHS + d];
}

// ---------------------------------------------------------------------------
extern "C" void kernel_launch(void* const* d_in, const int* in_sizes, int n_in,
                              void* d_out, int out_size, void* d_ws, size_t ws_size,
                              hipStream_t stream) {
  const float* seq    = (const float*)d_in[0];   // (4,1024,768)
  const float* att    = (const float*)d_in[1];   // (4,12,1024,1024)
  const int*   starts = (const int*)  d_in[2];   // (4,42,4)
  const int*   hts    = (const int*)  d_in[3];   // (4,1764,2)

  float* out = (float*)d_out;
  const size_t BLK = (size_t)NB * NR * NHS;      // 5,419,008 per output tensor
  float* hss = out;
  float* rss = out + BLK;
  float* tss = out + 2 * BLK;

  // workspace layout (bytes)
  char* ws = (char*)d_ws;
  float*    eemb = (float*)ws;                                   // 516,096 B
  float*    eatt = (float*)(ws + 516096);                        // 8,257,536 B
  _Float16* seqT = (_Float16*)(ws + 516096 + 8257536);           // 6,291,456 B
  _Float16* htA  = (_Float16*)(ws + 516096 + 8257536 + 6291456); // 14,450,688 B

  k_e_emb <<<NB * NE,               256, 0, stream>>>(seq, starts, eemb);
  k_seqT  <<<(NB * NHS * NC) / 256, 256, 0, stream>>>(seq, seqT);
  k_e_att <<<NB * NE * NH,          256, 0, stream>>>(att, starts, eatt);
  k_ht_att<<<NB * NR,               256, 0, stream>>>(eatt, hts, htA);
  // 4 docs * 111*12 (16x64) tiles = 5328 waves; 8 waves / 256-thread block
  k_gemm_rs<<<(NB * MT * NT64) / 8, 256, 0, stream>>>(htA, seqT, rss);
  k_gather_ht<<<(unsigned)(BLK / 256), 256, 0, stream>>>(eemb, hts, hss, tss);
}
